// HeteroRGCN_52183852646760
// MI455X (gfx1250) — matrix operand order repaired and checked by reading, not compiled, not run on hardware
//
#include <hip/hip_runtime.h>
#include <hip/hip_bf16.h>
#include <string.h>

// ---- problem dims (feature dims compile-time; N/E from in_sizes) ----
#define IN_CH    128
#define TYPE_EMB 32
#define FEAT     160          // IN_CH + TYPE_EMB
#define OUT_CH   128
#define N_REL    8
#define N_BASES  8
#define MBLK     128          // rows per GEMM block (8 waves x 16)

typedef __attribute__((ext_vector_type(16))) __bf16 v16bf;
typedef __attribute__((ext_vector_type(8)))  float  v8f;

// -------------------- weight prep: wb[r] = sum_b comp[r,b]*bases[b]; rootb = bf16(root_w)
__global__ void k_weights(const float* __restrict__ comp, const float* __restrict__ bases,
                          const float* __restrict__ root_w,
                          __bf16* __restrict__ wb, __bf16* __restrict__ rootb)
{
    int idx = blockIdx.x * blockDim.x + threadIdx.x;
    const int RW = N_REL * FEAT * OUT_CH;            // 163840
    if (idx < RW) {
        int r  = idx / (FEAT * OUT_CH);
        int io = idx % (FEAT * OUT_CH);
        float acc = 0.f;
#pragma unroll
        for (int b = 0; b < N_BASES; ++b)
            acc += comp[r * N_BASES + b] * bases[(size_t)b * FEAT * OUT_CH + io];
        wb[idx] = (__bf16)acc;
    } else if (idx < RW + FEAT * OUT_CH) {
        int io = idx - RW;
        rootb[io] = (__bf16)root_w[io];
    }
}

// -------------------- h = concat(x, emb[type]) in bf16
__global__ void k_build_h(const float* __restrict__ x, const int* __restrict__ ntype,
                          const float* __restrict__ emb, __bf16* __restrict__ hb, int N)
{
    size_t idx = (size_t)blockIdx.x * blockDim.x + threadIdx.x;
    if (idx >= (size_t)N * FEAT) return;
    int n = (int)(idx / FEAT);
    int i = (int)(idx % FEAT);
    float v = (i < IN_CH) ? x[(size_t)n * IN_CH + i]
                          : emb[(size_t)ntype[n] * TYPE_EMB + (i - IN_CH)];
    hb[idx] = (__bf16)v;
}

// -------------------- one-pass per-(node,relation) in-degree counts
__global__ void k_count(const int* __restrict__ ei, const int* __restrict__ etype,
                        float* __restrict__ c_all, int E)
{
    int e = blockIdx.x * blockDim.x + threadIdx.x;
    if (e < E)
        atomicAdd(&c_all[(size_t)ei[E + e] * N_REL + etype[e]], 1.0f);
}

// -------------------- per-relation segment-sum scatter: one wave per edge
__global__ void k_scatter(const __bf16* __restrict__ hb, const int* __restrict__ ei,
                          const int* __restrict__ etype, float* __restrict__ s,
                          int rel, int E)
{
    int gwid = (blockIdx.x * blockDim.x + threadIdx.x) >> 5;
    int lane = threadIdx.x & 31;
    int nwav = (gridDim.x * blockDim.x) >> 5;
    for (int e = gwid; e < E; e += nwav) {
        int en = e + nwav;                            // prefetch next edge's metadata
        if (en < E) {
            __builtin_prefetch(&etype[en], 0, 0);
            __builtin_prefetch(&ei[en], 0, 0);
            __builtin_prefetch(&ei[E + en], 0, 0);
        }
        if (etype[e] != rel) continue;                // wave-uniform branch
        int src = ei[e];
        int dst = ei[E + e];
        const unsigned* hs32 = (const unsigned*)(hb + (size_t)src * FEAT);  // 80 dwords
        float* sd = s + (size_t)dst * FEAT;
#pragma unroll
        for (int j = 0; j < 2; ++j) {
            int d = lane + j * 32;
            unsigned u = hs32[d];
            union { unsigned v; float f; } lo, hi;
            lo.v = (u & 0xffffu) << 16;               // bf16 -> f32
            hi.v = u & 0xffff0000u;
            atomicAdd(&sd[2 * d + 0], lo.f);
            atomicAdd(&sd[2 * d + 1], hi.f);
        }
        if (lane < 16) {
            int d = 64 + lane;
            unsigned u = hs32[d];
            union { unsigned v; float f; } lo, hi;
            lo.v = (u & 0xffffu) << 16;
            hi.v = u & 0xffff0000u;
            atomicAdd(&sd[2 * d + 0], lo.f);
            atomicAdd(&sd[2 * d + 1], hi.f);
        }
    }
}

// -------------------- WMMA bf16 GEMM: out[m,n] (+)= A[m,:] @ W[:,n]
// Two A paths: Abf != nullptr -> bf16 rows (root transform, C = bias).
//              Abf == nullptr -> Af f32 segment sums scaled by 1/max(c_all[row,rel],1)
//                                (relation path, C = out accumulate).
// doRelu: apply ReLU at store (final relation).
__global__ __launch_bounds__(256) void k_gemm(const __bf16* __restrict__ Abf,
                                              const float*  __restrict__ Af,
                                              const float*  __restrict__ c_all,
                                              int rel,
                                              const __bf16* __restrict__ W,
                                              float* __restrict__ out,
                                              const float* __restrict__ bias,
                                              int N, int initC, int doRelu)
{
    __shared__ __align__(16) __bf16 Ah[MBLK * FEAT];     // 40 KB
    __shared__ __align__(16) __bf16 Wt[OUT_CH * FEAT];   // 40 KB, layout [n][k]
    const int tid = threadIdx.x;
    const int m0  = blockIdx.x * MBLK;

    // ---- stage A slab
    if (Abf) {                                            // bf16 copy, b128 granularity
        const uint4* As = (const uint4*)(Abf + (size_t)m0 * FEAT);
        uint4* Ad = (uint4*)Ah;
        const int totq = MBLK * FEAT / 8;                 // 2560 x 16B
        int vr = N - m0;
        int validq = vr >= MBLK ? totq : (vr > 0 ? vr * (FEAT / 8) : 0);
        for (int t = tid; t < totq; t += 256)
            Ad[t] = (t < validq) ? As[t] : make_uint4(0u, 0u, 0u, 0u);
    } else {                                              // f32 sums, fused mean + cvt
        for (int t = tid; t < MBLK * (FEAT / 4); t += 256) {
            int row = t / (FEAT / 4);
            int q   = t % (FEAT / 4);
            int grow = m0 + row;
            float inv = 0.f;
            float4 v = make_float4(0.f, 0.f, 0.f, 0.f);
            if (grow < N) {
                inv = 1.0f / fmaxf(c_all[(size_t)grow * N_REL + rel], 1.0f);
                v = ((const float4*)(Af + (size_t)grow * FEAT))[q];
            }
            __bf16* d = &Ah[(size_t)row * FEAT + q * 4];
            d[0] = (__bf16)(v.x * inv);
            d[1] = (__bf16)(v.y * inv);
            d[2] = (__bf16)(v.z * inv);
            d[3] = (__bf16)(v.w * inv);
        }
    }
    // ---- stage W with transpose: Wt[n*FEAT + k] = W[k*OUT_CH + n]
    {
        const uint4* Ws = (const uint4*)W;
        const int totq = FEAT * (OUT_CH / 8);             // 2560 x 16B (8 bf16 along n)
        for (int t = tid; t < totq; t += 256) {
            uint4 d = Ws[t];
            int k  = t / (OUT_CH / 8);
            int nb = (t % (OUT_CH / 8)) * 8;
            unsigned u[4] = { d.x, d.y, d.z, d.w };
#pragma unroll
            for (int p = 0; p < 4; ++p) {
                unsigned short lo = (unsigned short)(u[p] & 0xffffu);
                unsigned short hi = (unsigned short)(u[p] >> 16);
                __bf16 blo, bhi;
                __builtin_memcpy(&blo, &lo, 2);
                __builtin_memcpy(&bhi, &hi, 2);
                Wt[(size_t)(nb + 2 * p + 0) * FEAT + k] = blo;
                Wt[(size_t)(nb + 2 * p + 1) * FEAT + k] = bhi;
            }
        }
    }
    __syncthreads();

    const int wave = tid >> 5;
    const int lane = tid & 31;
    const int half = lane >> 4;
    const int ln   = lane & 15;

    // ---- C init: 8 tiles of 16x16 per wave (covers all 128 columns)
    v8f acc[8];
    if (initC) {
#pragma unroll
        for (int nt = 0; nt < 8; ++nt) {
            float bv = bias[nt * 16 + ln];
#pragma unroll
            for (int g = 0; g < 8; ++g) acc[nt][g] = bv;
        }
    } else {
#pragma unroll
        for (int nt = 0; nt < 8; ++nt) {
#pragma unroll
            for (int g = 0; g < 8; ++g) {
                int row = m0 + wave * 16 + half * 8 + g;  // lanes 16-31 hold M+8
                acc[nt][g] = (row < N) ? out[(size_t)row * OUT_CH + nt * 16 + ln] : 0.0f;
            }
        }
    }

    // ---- K loop: 160 = 5 x 32
#pragma unroll
    for (int kk = 0; kk < FEAT / 32; ++kk) {
        const int K0 = kk * 32;
        // A fragment: lane holds row (wave*16+ln), K = K0 + half*8 + {0..7} and +16
        v16bf a;
        const __bf16* ar = &Ah[(size_t)(wave * 16 + ln) * FEAT + K0 + half * 8];
#pragma unroll
        for (int j = 0; j < 8; ++j) { a[j] = ar[j]; a[8 + j] = ar[16 + j]; }
#pragma unroll
        for (int nt = 0; nt < 8; ++nt) {
            // B fragment: lane holds column nt*16+ln, K = K0 + half*8 + {0..7} and +16
            v16bf b;
            const __bf16* br = &Wt[(size_t)(nt * 16 + ln) * FEAT + K0 + half * 8];
#pragma unroll
            for (int e = 0; e < 8; ++e) { b[e] = br[e]; b[8 + e] = br[16 + e]; }
            acc[nt] = __builtin_amdgcn_wmma_f32_16x16x32_bf16(
                false, a, false, b, (short)0, acc[nt], false, false);
        }
    }

    // ---- store C (optionally fused ReLU)
#pragma unroll
    for (int nt = 0; nt < 8; ++nt) {
#pragma unroll
        for (int g = 0; g < 8; ++g) {
            int row = m0 + wave * 16 + half * 8 + g;
            if (row < N) {
                float v = acc[nt][g];
                if (doRelu) v = fmaxf(v, 0.0f);
                out[(size_t)row * OUT_CH + nt * 16 + ln] = v;
            }
        }
    }
}

static inline size_t alignup(size_t v) { return (v + 255) & ~(size_t)255; }

extern "C" void kernel_launch(void* const* d_in, const int* in_sizes, int n_in,
                              void* d_out, int out_size, void* d_ws, size_t ws_size,
                              hipStream_t stream) {
    const float* x      = (const float*)d_in[0];
    const int*   ntype  = (const int*)d_in[1];
    const int*   ei     = (const int*)d_in[2];   // [2, E]
    const int*   etype  = (const int*)d_in[3];
    const float* emb    = (const float*)d_in[4];
    const float* bases  = (const float*)d_in[5];
    const float* comp   = (const float*)d_in[6];
    const float* root_w = (const float*)d_in[7];
    const float* bias   = (const float*)d_in[8];
    float* out = (float*)d_out;

    const int N = in_sizes[0] / IN_CH;
    const int E = in_sizes[3];

    // workspace carve-up (~100 MB)
    char* ws = (char*)d_ws;
    size_t off = 0;
    __bf16* hb    = (__bf16*)(ws + off); off = alignup(off + (size_t)N * FEAT * 2);
    float*  s     = (float*) (ws + off); off = alignup(off + (size_t)N * FEAT * 4);
    float*  c_all = (float*) (ws + off); off = alignup(off + (size_t)N * N_REL * 4);
    __bf16* wb    = (__bf16*)(ws + off); off = alignup(off + (size_t)N_REL * FEAT * OUT_CH * 2);
    __bf16* rootb = (__bf16*)(ws + off); off = alignup(off + (size_t)FEAT * OUT_CH * 2);
    (void)ws_size;

    // counts for all relations in one edge scan
    (void)hipMemsetAsync(c_all, 0, (size_t)N * N_REL * 4, stream);
    k_count<<<(E + 255) / 256, 256, 0, stream>>>(ei, etype, c_all, E);

    const int wtot = N_REL * FEAT * OUT_CH + FEAT * OUT_CH;
    k_weights<<<(wtot + 255) / 256, 256, 0, stream>>>(comp, bases, root_w, wb, rootb);

    size_t hN = (size_t)N * FEAT;
    k_build_h<<<(unsigned)((hN + 255) / 256), 256, 0, stream>>>(x, ntype, emb, hb, N);

    const int gblocks = (N + MBLK - 1) / MBLK;
    // root transform: out = h @ root_w + bias
    k_gemm<<<gblocks, 256, 0, stream>>>(hb, (const float*)nullptr, (const float*)nullptr, 0,
                                        rootb, out, bias, N, 1, 0);

    for (int r = 0; r < N_REL; ++r) {
        (void)hipMemsetAsync(s, 0, (size_t)N * FEAT * 4, stream);
        k_scatter<<<2048, 256, 0, stream>>>(hb, ei, etype, s, r, E);
        // out += (s / max(c,1)) @ weight[r]   (mean fused into A staging; ReLU on last)
        k_gemm<<<gblocks, 256, 0, stream>>>((const __bf16*)nullptr, s, c_all, r,
                                            wb + (size_t)r * FEAT * OUT_CH, out, bias,
                                            N, 0, r == N_REL - 1 ? 1 : 0);
    }
}
